// MaskedSelfAttention_46102178955918
// MI455X (gfx1250) — matrix operand order repaired and checked by reading, not compiled, not run on hardware
//
#include <hip/hip_runtime.h>

#define S_LEN  2048
#define EMB    1024
#define NHEAD  16
#define HDIM   64
#define NBATCH 2
#define MROWS  (NBATCH * S_LEN)   /* 4096 */
#define LN_EPS 1e-5f

typedef __bf16 bf16;
typedef __attribute__((ext_vector_type(16))) __bf16 v16bf;
typedef __attribute__((ext_vector_type(8)))  float  v8f;

// ---------------------------------------------------------------------------
// D = A(16x32 bf16) * B(32x16 bf16) + C(16x16 f32)
// ---------------------------------------------------------------------------
__device__ __forceinline__ v8f wmma_bf16(v16bf a, v16bf b, v8f c) {
  return __builtin_amdgcn_wmma_f32_16x16x32_bf16(
      /*neg_a=*/false, a, /*neg_b=*/false, b,
      /*c_mod=*/(short)0, c, /*reuse_a=*/false, /*reuse_b=*/false);
}

// ---------------------------------------------------------------------------
// Fragment loaders (contiguous-k layouts used by this file, wave32).
// A (16xK=32, 16-bit): lanes 0-15 -> M=0..15 ; lane-half selects K sub-range.
//   slot i in v16bf: k = 16*(lane>=16) + i
// B (K=32 x 16, 16-bit): lane%16 -> N ; lane-half selects K half;
//   slot i -> k = 16*(lane>=16) + i
// C/D (16x16 f32): vgpr j -> M = j + 8*(lane>=16), N = lane%16
// ---------------------------------------------------------------------------
__device__ __forceinline__ v16bf load_A_frag_bf16(const bf16* src, int ld) {
  int lane = threadIdx.x & 31;
  int m = lane & 15, half = lane >> 4;
  const bf16* p = src + (size_t)m * ld + half * 16;
  v16bf a;
#pragma unroll
  for (int i = 0; i < 8; ++i) a[i] = p[i];
#pragma unroll
  for (int i = 0; i < 8; ++i) a[8 + i] = p[8 + i];
  return a;
}

// B[k][n] = src[n*ld + k]  (transposed source: W row-major N x K, or K^T)
__device__ __forceinline__ v16bf load_B_fragT_bf16(const bf16* src, int ld) {
  int lane = threadIdx.x & 31;
  int n = lane & 15, kh = (lane >> 4) * 16;
  const bf16* p = src + (size_t)n * ld + kh;
  v16bf b;
#pragma unroll
  for (int i = 0; i < 16; ++i) b[i] = p[i];
  return b;
}

// B[k][n] = src[k*ld + n]  (row-major K x N source, e.g. V)
__device__ __forceinline__ v16bf load_B_frag_bf16(const bf16* src, int ld) {
  int lane = threadIdx.x & 31;
  int n = lane & 15, kh = (lane >> 4) * 16;
  v16bf b;
#pragma unroll
  for (int i = 0; i < 16; ++i) b[i] = src[(size_t)(kh + i) * ld + n];
  return b;
}

// ---------------------------------------------------------------------------
// Kernel 0: one-time fp32 -> bf16 conversion (weights reused 256x by GEMMs,
// so convert once instead of per-block in the hot loops).
// ---------------------------------------------------------------------------
__global__ void __launch_bounds__(256)
cvt_kernel(const float* __restrict__ src, bf16* __restrict__ dst, int n) {
  int i = (blockIdx.x * 256 + threadIdx.x) * 4;
  if (i < n) {
    float4 f = *(const float4*)(src + i);
    dst[i + 0] = (bf16)f.x;
    dst[i + 1] = (bf16)f.y;
    dst[i + 2] = (bf16)f.z;
    dst[i + 3] = (bf16)f.w;
  }
}

// ---------------------------------------------------------------------------
// Kernel 1: fused Q/K/V projections. y = x @ W.T + b  (torch Linear)
// grid = (MROWS/16, 2, 3), block = 256 (8 waves).
// Each wave computes a 16x64 output tile (one full head width) -> 4 WMMAs
// per K-step reusing one A fragment. The 16x32 bf16 A tile (shared by all
// 8 waves) is staged into LDS per K-step with async-to-LDS copies.
// ---------------------------------------------------------------------------
__global__ void __launch_bounds__(256)
qkv_proj_kernel(const bf16* __restrict__ Xb,
                const bf16* __restrict__ Wqb, const float* __restrict__ bq,
                const bf16* __restrict__ Wkb, const float* __restrict__ bk,
                const bf16* __restrict__ Wvb, const float* __restrict__ bv,
                bf16* __restrict__ qw, bf16* __restrict__ kw, bf16* __restrict__ vw) {
  __shared__ __align__(16) bf16 Atile[16 * 32];

  int which = blockIdx.z;
  const bf16* W     = which == 0 ? Wqb : (which == 1 ? Wkb : Wvb);
  const float* bias = which == 0 ? bq : (which == 1 ? bk : bv);
  bf16* dst         = which == 0 ? qw : (which == 1 ? kw : vw);

  int tid = threadIdx.x;
  int wave = tid >> 5, lane = tid & 31;
  int mt = blockIdx.x;
  int ng = blockIdx.y * 8 + wave;          // 64-wide column group == head id
  const bf16* Arow = Xb + (size_t)mt * 16 * EMB;
  const bf16* Brow = W + (size_t)ng * 64 * EMB;

  unsigned int lds_base = (unsigned int)(uintptr_t)(&Atile[0]);

  v8f acc[4] = {};
  for (int k = 0; k < EMB; k += 32) {
    // Stage A tile (16 rows x 32 cols bf16 = 1KB) into LDS: 64 lanes x 16B,
    // ASYNCcnt-tracked async copies (waves 0-1 issue; uniform per wave).
    if (tid < 64) {
      int row = tid >> 2, col = (tid & 3) * 8;
      unsigned long long g =
          (unsigned long long)(uintptr_t)(Arow + (size_t)row * EMB + k + col);
      unsigned int l = lds_base + (unsigned int)tid * 16u;
      asm volatile("global_load_async_to_lds_b128 %0, %1, off"
                   :: "v"(l), "v"(g) : "memory");
    }
    asm volatile("s_wait_asynccnt 0" ::: "memory");
    __syncthreads();

    v16bf a = load_A_frag_bf16(Atile, 32);
    __builtin_prefetch(Brow + k + 512, 0, 1);
#pragma unroll
    for (int t = 0; t < 4; ++t)
      acc[t] = wmma_bf16(a, load_B_fragT_bf16(Brow + (size_t)t * 16 * EMB + k, EMB),
                         acc[t]);
    __syncthreads();
  }

  int half = lane >> 4, ln = lane & 15;
#pragma unroll
  for (int t = 0; t < 4; ++t) {
    int n_g = ng * 64 + t * 16 + ln;
    float bval = bias[n_g];
#pragma unroll
    for (int j = 0; j < 8; ++j) {
      int m_g = mt * 16 + j + half * 8;
      int b = m_g >> 11, s = m_g & 2047;
      dst[(((size_t)(b * NHEAD + ng)) * S_LEN + s) * HDIM + t * 16 + ln] =
          (bf16)(acc[t][j] + bval);
    }
  }
}

// ---------------------------------------------------------------------------
// Kernel 2: causal flash attention. One 16-row Q tile per wave.
// grid = (S/16/4, B*H), block = 128 (4 waves).
// ---------------------------------------------------------------------------
__global__ void __launch_bounds__(128)
attn_kernel(const bf16* __restrict__ qw, const bf16* __restrict__ kw,
            const bf16* __restrict__ vw, bf16* __restrict__ aw) {
  __shared__ bf16 Pbuf[4][16 * 32];

  int wave = threadIdx.x >> 5, lane = threadIdx.x & 31;
  int half = lane >> 4, ln = lane & 15;
  int qt = blockIdx.x * 4 + wave;
  int bh = blockIdx.y;
  int b = bh >> 4, h = bh & 15;

  const bf16* Qbh = qw + (size_t)bh * S_LEN * HDIM;
  const bf16* Kbh = kw + (size_t)bh * S_LEN * HDIM;
  const bf16* Vbh = vw + (size_t)bh * S_LEN * HDIM;
  bf16* pb = Pbuf[wave];

  v16bf qa0 = load_A_frag_bf16(Qbh + (size_t)qt * 16 * HDIM, HDIM);
  v16bf qa1 = load_A_frag_bf16(Qbh + (size_t)qt * 16 * HDIM + 32, HDIM);

  v8f o0 = {}, o1 = {}, o2 = {}, o3 = {};
  float mrow[8], lrow[8], scj[8];
#pragma unroll
  for (int j = 0; j < 8; ++j) { mrow[j] = -__builtin_inff(); lrow[j] = 0.f; }

  int nkb = (qt * 16 + 15) / 32 + 1;  // causal: key blocks up to the diagonal
  for (int kb = 0; kb < nkb; ++kb) {
    int ktbase = kb * 32;
    const bf16* K0 = Kbh + (size_t)ktbase * HDIM;
    const bf16* K1 = K0 + 16 * HDIM;

    // S = Q(16x64) * K^T(64x32), split into two 16x16 N-tiles, K split 2x32
    v8f s0 = {}, s1 = {};
    s0 = wmma_bf16(qa0, load_B_fragT_bf16(K0, HDIM), s0);
    s0 = wmma_bf16(qa1, load_B_fragT_bf16(K0 + 32, HDIM), s0);
    s1 = wmma_bf16(qa0, load_B_fragT_bf16(K1, HDIM), s1);
    s1 = wmma_bf16(qa1, load_B_fragT_bf16(K1 + 32, HDIM), s1);

    bool needMask = (ktbase + 31) > (qt * 16);
#pragma unroll
    for (int j = 0; j < 8; ++j) {
      float e0 = s0[j] * 0.125f, e1 = s1[j] * 0.125f;  // 1/sqrt(64)
      if (needMask) {
        int qr = qt * 16 + j + half * 8;
        if (ktbase + ln > qr)      e0 = -__builtin_inff();
        if (ktbase + 16 + ln > qr) e1 = -__builtin_inff();
      }
      // row max across the 16 lanes of this half (xor stays within half)
      float rm = fmaxf(e0, e1);
      rm = fmaxf(rm, __shfl_xor(rm, 1, 32));
      rm = fmaxf(rm, __shfl_xor(rm, 2, 32));
      rm = fmaxf(rm, __shfl_xor(rm, 4, 32));
      rm = fmaxf(rm, __shfl_xor(rm, 8, 32));
      float mnew = fmaxf(mrow[j], rm);
      float corr = __expf(mrow[j] - mnew);
      float p0 = __expf(e0 - mnew), p1 = __expf(e1 - mnew);
      float rs = p0 + p1;
      rs += __shfl_xor(rs, 1, 32);
      rs += __shfl_xor(rs, 2, 32);
      rs += __shfl_xor(rs, 4, 32);
      rs += __shfl_xor(rs, 8, 32);
      lrow[j] = lrow[j] * corr + rs;
      mrow[j] = mnew;
      scj[j] = corr;
      // stash P in LDS (D-layout -> row-major) for re-read in A-layout
      pb[(j + half * 8) * 32 + ln]      = (bf16)p0;
      pb[(j + half * 8) * 32 + 16 + ln] = (bf16)p1;
    }
#pragma unroll
    for (int j = 0; j < 8; ++j) {
      o0[j] *= scj[j]; o1[j] *= scj[j]; o2[j] *= scj[j]; o3[j] *= scj[j];
    }
    // ensure DS stores complete/order before same-wave DS reads of Pbuf
    asm volatile("s_wait_dscnt 0" ::: "memory");
    v16bf pa = load_A_frag_bf16(pb, 32);

    const bf16* Vt = Vbh + (size_t)ktbase * HDIM;
    o0 = wmma_bf16(pa, load_B_frag_bf16(Vt + 0,  HDIM), o0);
    o1 = wmma_bf16(pa, load_B_frag_bf16(Vt + 16, HDIM), o1);
    o2 = wmma_bf16(pa, load_B_frag_bf16(Vt + 32, HDIM), o2);
    o3 = wmma_bf16(pa, load_B_frag_bf16(Vt + 48, HDIM), o3);
  }

#pragma unroll
  for (int j = 0; j < 8; ++j) {
    float inv = 1.0f / lrow[j];
    int q = qt * 16 + j + half * 8;
    size_t base = (((size_t)(b * S_LEN + q)) * NHEAD + h) * HDIM + ln;
    aw[base + 0]  = (bf16)(o0[j] * inv);
    aw[base + 16] = (bf16)(o1[j] * inv);
    aw[base + 32] = (bf16)(o2[j] * inv);
    aw[base + 48] = (bf16)(o3[j] * inv);
  }
}

// ---------------------------------------------------------------------------
// Kernel 3: out-projection + bias + residual. grid = (MROWS/16, 2), block 256.
// 16x64 tile per wave (4 WMMAs per K-step, shared A fragment).
// ---------------------------------------------------------------------------
__global__ void __launch_bounds__(256)
out_proj_kernel(const bf16* __restrict__ aw, const bf16* __restrict__ Wob,
                const float* __restrict__ bo, const float* __restrict__ X,
                float* __restrict__ hw) {
  int wave = threadIdx.x >> 5, lane = threadIdx.x & 31;
  int mt = blockIdx.x, ng = blockIdx.y * 8 + wave;
  const bf16* Arow = aw + (size_t)mt * 16 * EMB;
  const bf16* Brow = Wob + (size_t)ng * 64 * EMB;

  v8f acc[4] = {};
  for (int k = 0; k < EMB; k += 32) {
    __builtin_prefetch(Brow + k + 512, 0, 1);
    v16bf a = load_A_frag_bf16(Arow + k, EMB);
#pragma unroll
    for (int t = 0; t < 4; ++t)
      acc[t] = wmma_bf16(a, load_B_fragT_bf16(Brow + (size_t)t * 16 * EMB + k, EMB),
                         acc[t]);
  }

  int half = lane >> 4, ln = lane & 15;
#pragma unroll
  for (int t = 0; t < 4; ++t) {
    int n_g = ng * 64 + t * 16 + ln;
    float bval = bo[n_g];
#pragma unroll
    for (int j = 0; j < 8; ++j) {
      int m_g = mt * 16 + j + half * 8;
      size_t idx = (size_t)m_g * EMB + n_g;
      hw[idx] = acc[t][j] + bval + X[idx];  // residual fused here (fp32 x)
    }
  }
}

// ---------------------------------------------------------------------------
// Kernel 4: row LayerNorm. grid = MROWS, block = 256.
// ---------------------------------------------------------------------------
__global__ void __launch_bounds__(256)
ln_kernel(const float* __restrict__ hbuf, const float* __restrict__ gamma,
          const float* __restrict__ beta, float* __restrict__ out) {
  __shared__ float r1[256];
  __shared__ float r2[256];
  int row = blockIdx.x, tid = threadIdx.x;
  const float* hr = hbuf + (size_t)row * EMB;
  float v[4];
  float s = 0.f, sq = 0.f;
#pragma unroll
  for (int r = 0; r < 4; ++r) {
    v[r] = hr[tid + 256 * r];
    s += v[r];
    sq += v[r] * v[r];
  }
  r1[tid] = s; r2[tid] = sq;
  __syncthreads();
  for (int off = 128; off > 0; off >>= 1) {
    if (tid < off) { r1[tid] += r1[tid + off]; r2[tid] += r2[tid + off]; }
    __syncthreads();
  }
  float mean = r1[0] * (1.0f / EMB);
  float var  = r2[0] * (1.0f / EMB) - mean * mean;
  float inv  = rsqrtf(var + LN_EPS);
#pragma unroll
  for (int r = 0; r < 4; ++r) {
    int i = tid + 256 * r;
    out[(size_t)row * EMB + i] = (v[r] - mean) * inv * gamma[i] + beta[i];
  }
}

// ---------------------------------------------------------------------------
// Launcher. Workspace layout (64 MB):
//   qw/kw/vw : 3 x 4096*1024 bf16   ([b,h,s,d])
//   aw       : 4096*1024 bf16       ([b,s,h*d] = rows of attention output)
//   hw       : 4096*1024 f32        (pre-LayerNorm h = proj + bias + x)
//   xb       : 4096*1024 bf16       (x converted once)
//   wqb/wkb/wvb/wob : 4 x 1024*1024 bf16 (weights converted once)
// ---------------------------------------------------------------------------
extern "C" void kernel_launch(void* const* d_in, const int* in_sizes, int n_in,
                              void* d_out, int out_size, void* d_ws, size_t ws_size,
                              hipStream_t stream) {
  (void)in_sizes; (void)n_in; (void)out_size; (void)ws_size;
  const float* x     = (const float*)d_in[0];
  const float* Wq    = (const float*)d_in[1];
  const float* bq    = (const float*)d_in[2];
  const float* Wk    = (const float*)d_in[3];
  const float* bk    = (const float*)d_in[4];
  const float* Wv    = (const float*)d_in[5];
  const float* bv    = (const float*)d_in[6];
  const float* Wo    = (const float*)d_in[7];
  const float* bo    = (const float*)d_in[8];
  const float* gamma = (const float*)d_in[9];
  const float* beta  = (const float*)d_in[10];
  float* out = (float*)d_out;

  size_t mat = (size_t)MROWS * EMB;      // 4M elements
  size_t wmat = (size_t)EMB * EMB;       // 1M elements
  bf16* qw  = (bf16*)d_ws;
  bf16* kw  = qw + mat;
  bf16* vw  = kw + mat;
  bf16* aw  = vw + mat;
  float* hw = (float*)(aw + mat);
  bf16* xb  = (bf16*)(hw + mat);
  bf16* wqb = xb + mat;
  bf16* wkb = wqb + wmat;
  bf16* wvb = wkb + wmat;
  bf16* wob = wvb + wmat;

  // One-time bf16 conversions (x + 4 weight matrices)
  cvt_kernel<<<dim3((int)(mat / 1024)),  dim3(256), 0, stream>>>(x,  xb,  (int)mat);
  cvt_kernel<<<dim3((int)(wmat / 1024)), dim3(256), 0, stream>>>(Wq, wqb, (int)wmat);
  cvt_kernel<<<dim3((int)(wmat / 1024)), dim3(256), 0, stream>>>(Wk, wkb, (int)wmat);
  cvt_kernel<<<dim3((int)(wmat / 1024)), dim3(256), 0, stream>>>(Wv, wvb, (int)wmat);
  cvt_kernel<<<dim3((int)(wmat / 1024)), dim3(256), 0, stream>>>(Wo, wob, (int)wmat);

  qkv_proj_kernel<<<dim3(MROWS / 16, EMB / 64 / 8, 3), dim3(256), 0, stream>>>(
      xb, wqb, bq, wkb, bk, wvb, bv, qw, kw, vw);
  attn_kernel<<<dim3(S_LEN / 16 / 4, NBATCH * NHEAD), dim3(128), 0, stream>>>(
      qw, kw, vw, aw);
  out_proj_kernel<<<dim3(MROWS / 16, EMB / 64 / 8), dim3(256), 0, stream>>>(
      aw, wob, bo, x, hw);
  ln_kernel<<<dim3(MROWS), dim3(256), 0, stream>>>(hw, gamma, beta, out);
}
